// HardAlign_74071005987588
// MI455X (gfx1250) — matrix operand order repaired and checked by reading, not compiled, not run on hardware
//
#include <hip/hip_runtime.h>

// Problem dims (fixed by reference)
#define BB 8
#define NN 4096
#define MM 4096
#define DD 128
#define LSTR 136   // LDS row stride in halves (272B): kills 16-way bank conflicts

typedef __attribute__((ext_vector_type(16))) __bf16 v16bf;
typedef __attribute__((ext_vector_type(8)))  float  v8f;
typedef int v4i_gcc __attribute__((vector_size(16)));   // matches async builtin param

union V16BF { v16bf v; unsigned short u[16]; unsigned int w[8]; };

// ---------------------------------------------------------------------------
// gfx1250 async global->LDS path (ASYNCcnt), guarded so either toolchain builds.
// ---------------------------------------------------------------------------
#if defined(__HIP_DEVICE_COMPILE__) && __has_builtin(__builtin_amdgcn_global_load_async_to_lds_b128)
#define HAVE_ASYNC_LDS 1
#else
#define HAVE_ASYNC_LDS 0
#endif

__device__ __forceinline__ void async_copy16(const void* g, void* l) {
#if HAVE_ASYNC_LDS
    __builtin_amdgcn_global_load_async_to_lds_b128(
        (__attribute__((address_space(1))) v4i_gcc*)g,   // global src (AS1)
        (__attribute__((address_space(3))) v4i_gcc*)l,   // LDS dst (AS3)
        0, 0);
#else
    *(uint4*)l = *(const uint4*)g;
#endif
}
__device__ __forceinline__ void wait_async_le8() {
#if HAVE_ASYNC_LDS
#if __has_builtin(__builtin_amdgcn_s_wait_asynccnt)
    __builtin_amdgcn_s_wait_asynccnt(8);
#else
    asm volatile("s_wait_asynccnt 0x8" ::: "memory");
#endif
#endif
}
__device__ __forceinline__ void wait_async_le0() {
#if HAVE_ASYNC_LDS
#if __has_builtin(__builtin_amdgcn_s_wait_asynccnt)
    __builtin_amdgcn_s_wait_asynccnt(0);
#else
    asm volatile("s_wait_asynccnt 0x0" ::: "memory");
#endif
#endif
}

// Split fp32 into bf16-hi (truncation) and bf16-lo (RNE of residual).
__device__ __forceinline__ void bf16_split(float x, unsigned short& hi, unsigned short& lo) {
    unsigned uh = __float_as_uint(x) & 0xffff0000u;
    hi = (unsigned short)(uh >> 16);
    float res = x - __uint_as_float(uh);
    unsigned ul = __float_as_uint(res);
    lo = (unsigned short)((ul + 0x7fffu + ((ul >> 16) & 1u)) >> 16);
}

// Load a 16-half B fragment from (16B-aligned) padded LDS as two b128 ops.
__device__ __forceinline__ void load_frag(const unsigned short* p, V16BF& f) {
    *(uint4*)&f.u[0] = *(const uint4*)(p);
    *(uint4*)&f.u[8] = *(const uint4*)(p + 8);
}

// ---------------------------------------------------------------------------
// Precompute kernels (ws path): bf16 hi/lo split of P, and deterministic p^2.
// ---------------------------------------------------------------------------
__global__ __launch_bounds__(256) void split_kernel(const float* __restrict__ P,
                                                    unsigned short* __restrict__ Phi,
                                                    unsigned short* __restrict__ Plo) {
    size_t gid = (size_t)blockIdx.x * 256 + threadIdx.x;   // float4 id
    const float4 x4 = ((const float4*)P)[gid];
    unsigned short h0,h1,h2,h3,l0,l1,l2,l3;
    bf16_split(x4.x, h0, l0); bf16_split(x4.y, h1, l1);
    bf16_split(x4.z, h2, l2); bf16_split(x4.w, h3, l3);
    uint2 hv, lv;
    hv.x = (unsigned)h0 | ((unsigned)h1 << 16); hv.y = (unsigned)h2 | ((unsigned)h3 << 16);
    lv.x = (unsigned)l0 | ((unsigned)l1 << 16); lv.y = (unsigned)l2 | ((unsigned)l3 << 16);
    ((uint2*)Phi)[gid] = hv;
    ((uint2*)Plo)[gid] = lv;
}

__global__ __launch_bounds__(256) void p2_kernel(const float* __restrict__ P,
                                                 float* __restrict__ P2) {
    size_t row = (size_t)blockIdx.x * 256 + threadIdx.x;   // 0 .. B*M-1
    const float4* rp = (const float4*)(P + row * DD);
    float s = 0.0f;
#pragma unroll
    for (int j = 0; j < DD / 4; ++j) {
        float4 v = rp[j];
        s += v.x*v.x + v.y*v.y + v.z*v.z + v.w*v.w;
    }
    P2[row] = s;
}

// ---------------------------------------------------------------------------
// Main kernel (ws path): bf16 prompt stream, double-buffered async LDS tiles.
// ---------------------------------------------------------------------------
__device__ __forceinline__ void issue_chunk_loads(const unsigned short* gH,
                                                  const unsigned short* gL,
                                                  unsigned short* lH,
                                                  unsigned short* lL, int t) {
    // 64 prompts x 128 halves = 1024 x 16B units; 256 threads x 4 units each.
    // Global is dense (row stride 128 halves); LDS rows padded to LSTR halves.
#pragma unroll
    for (int i = 0; i < 4; ++i) {
        int u   = t + (i << 8);     // 16B-unit id, 0..1023
        int row = u >> 4;           // 16 units per row
        int ur  = u & 15;
        int lo  = row * LSTR + ur * 8;   // half offset in LDS
        async_copy16(gH + u * 8, lH + lo);
        async_copy16(gL + u * 8, lL + lo);
    }
}

__global__ __launch_bounds__(256) void
nn_wmma_pre_kernel(const unsigned short* __restrict__ Phi,   // [B,M,D] bf16-hi
                   const unsigned short* __restrict__ Plo,   // [B,M,D] bf16-lo
                   const float* __restrict__ P2,             // [B,M]
                   const float* __restrict__ Pf,             // [B,M,D] fp32 (gather)
                   const float* __restrict__ Q,              // [B,N,D] fp32
                   float* __restrict__ out) {                // [B,N,D]
    __shared__ __align__(16) unsigned short ldsHi[2][64 * LSTR];  // 2 x 17 KB
    __shared__ __align__(16) unsigned short ldsLo[2][64 * LSTR];  // 2 x 17 KB
    __shared__ int ldsIdx[128];

    const int t     = threadIdx.x;
    const int lane  = t & 31;
    const int wave  = t >> 5;
    const int row16 = lane & 15;
    const int half  = lane >> 4;
    const int b     = blockIdx.y;
    const int qBaseWG = blockIdx.x * 128;

    const unsigned short* PhiB = Phi + (size_t)b * MM * DD;
    const unsigned short* PloB = Plo + (size_t)b * MM * DD;
    const float* P2b = P2 + (size_t)b * MM;
    const float* Pfb = Pf + (size_t)b * MM * DD;
    const float* Qb  = Q  + (size_t)b * NN * DD;

    // Resident A fragments: this wave's 16 queries, bf16 hi/lo, contiguous k order (k = 16*half + i).
    V16BF ahi[4], alo[4];
    {
        const int qrow = qBaseWG + wave * 16 + row16;
        const float* qp = Qb + (size_t)qrow * DD;
#pragma unroll
        for (int ks = 0; ks < 4; ++ks) {
#pragma unroll
            for (int i = 0; i < 16; ++i) {
                int v = i >> 1, slot = i & 1;
                int k = ks * 32 + (half << 4) + (v << 1) + slot;
                bf16_split(qp[k], ahi[ks].u[i], alo[ks].u[i]);
            }
        }
    }

    float best[8];
    int   bidx[8];
#pragma unroll
    for (int r = 0; r < 8; ++r) { best[r] = __builtin_inff(); bidx[r] = 0; }

    // Prologue: start chunk 0 into buffer 0; prefetch chunk-0 p^2 to registers.
    issue_chunk_loads(PhiB, PloB, ldsHi[0], ldsLo[0], t);
    float p2v[4];
#pragma unroll
    for (int st = 0; st < 4; ++st) p2v[st] = P2b[(st << 4) + row16];

    for (int c = 0; c < MM / 64; ++c) {
        const int buf   = c & 1;
        const int chunk = c << 6;
        float p2n[4] = {0.f, 0.f, 0.f, 0.f};
        if (c + 1 < MM / 64) {   // prefetch next chunk into the other buffer
            issue_chunk_loads(PhiB + (size_t)(chunk + 64) * DD,
                              PloB + (size_t)(chunk + 64) * DD,
                              ldsHi[buf ^ 1], ldsLo[buf ^ 1], t);
#pragma unroll
            for (int st = 0; st < 4; ++st)           // p^2 prefetch hides behind WMMAs
                p2n[st] = P2b[chunk + 64 + (st << 4) + row16];
            wait_async_le8();    // this wave's chunk-c loads are done
        } else {
            wait_async_le0();
        }
        __syncthreads();         // all waves' chunk-c loads landed in LDS

#pragma unroll
        for (int st = 0; st < 4; ++st) {
            const int col  = (st << 4) + row16;
            const int koff = col * LSTR + (half << 4);
            // Two independent accumulation chains (k-step parity) for WMMA ILP.
            v8f c0 = {0.f,0.f,0.f,0.f,0.f,0.f,0.f,0.f};
            v8f c1 = {0.f,0.f,0.f,0.f,0.f,0.f,0.f,0.f};
#pragma unroll
            for (int ks = 0; ks < 4; ++ks) {
                V16BF bh, bl;
                load_frag(&ldsHi[buf][koff + (ks << 5)], bh);
                load_frag(&ldsLo[buf][koff + (ks << 5)], bl);
                if (ks & 1) {
                    c1 = __builtin_amdgcn_wmma_f32_16x16x32_bf16(false, ahi[ks].v, false, bh.v,
                                                                 (short)0, c1, false, false);
                    c1 = __builtin_amdgcn_wmma_f32_16x16x32_bf16(false, ahi[ks].v, false, bl.v,
                                                                 (short)0, c1, false, false);
                    c1 = __builtin_amdgcn_wmma_f32_16x16x32_bf16(false, alo[ks].v, false, bh.v,
                                                                 (short)0, c1, false, false);
                } else {
                    c0 = __builtin_amdgcn_wmma_f32_16x16x32_bf16(false, ahi[ks].v, false, bh.v,
                                                                 (short)0, c0, false, false);
                    c0 = __builtin_amdgcn_wmma_f32_16x16x32_bf16(false, ahi[ks].v, false, bl.v,
                                                                 (short)0, c0, false, false);
                    c0 = __builtin_amdgcn_wmma_f32_16x16x32_bf16(false, alo[ks].v, false, bh.v,
                                                                 (short)0, c0, false, false);
                }
            }
            const int mIdx = chunk + (st << 4) + row16;
#pragma unroll
            for (int r = 0; r < 8; ++r) {
                float s = __builtin_fmaf(-2.0f, c0[r] + c1[r], p2v[st]);  // p^2 - 2*cross
                if (s < best[r]) { best[r] = s; bidx[r] = mIdx; }
            }
        }
        __syncthreads();         // compute done before next prefetch reuses buf
#pragma unroll
        for (int st = 0; st < 4; ++st) p2v[st] = p2n[st];
    }

    // Cross-lane argmin within each 16-lane half group (tie-break: lower index).
#pragma unroll
    for (int r = 0; r < 8; ++r) {
        float bv = best[r]; int bi = bidx[r];
#pragma unroll
        for (int m = 1; m <= 8; m <<= 1) {
            float ov = __shfl_xor(bv, m, 32);
            int   oi = __shfl_xor(bi, m, 32);
            if (ov < bv || (ov == bv && oi < bi)) { bv = ov; bi = oi; }
        }
        best[r] = bv; bidx[r] = bi;
    }
    if (row16 == 0) {
#pragma unroll
        for (int r = 0; r < 8; ++r)
            ldsIdx[wave * 16 + (half << 3) + r] = bidx[r];
    }
    __syncthreads();

    // Gather winning prompt rows from fp32 prompts.
    {
        const int q = t >> 1;
        const int h = t & 1;
        const int idx = ldsIdx[q];
        const float4* src = (const float4*)(Pfb + (size_t)idx * DD + h * 64);
        float4* dst = (float4*)(out + ((size_t)b * NN + qBaseWG + q) * DD + h * 64);
#pragma unroll
        for (int j = 0; j < 16; ++j) dst[j] = src[j];
    }
}

// ---------------------------------------------------------------------------
// Fallback (no-ws) kernel: in-kernel conversion, deterministic p^2 (no atomics).
// ---------------------------------------------------------------------------
__global__ __launch_bounds__(256) void
nn_gather_kernel(const float* __restrict__ P,
                 const float* __restrict__ Q,
                 float* __restrict__ out) {
    __shared__ __align__(16) unsigned short ldsHi[64 * LSTR];
    __shared__ __align__(16) unsigned short ldsLo[64 * LSTR];
    __shared__ float ldsP2[64];
    __shared__ int   ldsIdx[128];

    const int t     = threadIdx.x;
    const int lane  = t & 31;
    const int wave  = t >> 5;
    const int row16 = lane & 15;
    const int half  = lane >> 4;
    const int b     = blockIdx.y;
    const int qBaseWG = blockIdx.x * 128;

    const float* Pb = P + (size_t)b * MM * DD;
    const float* Qb = Q + (size_t)b * NN * DD;

    V16BF ahi[4], alo[4];
    {
        const int qrow = qBaseWG + wave * 16 + row16;
        const float* qp = Qb + (size_t)qrow * DD;
#pragma unroll
        for (int ks = 0; ks < 4; ++ks) {
#pragma unroll
            for (int i = 0; i < 16; ++i) {
                int v = i >> 1, slot = i & 1;
                int k = ks * 32 + (half << 4) + (v << 1) + slot;
                bf16_split(qp[k], ahi[ks].u[i], alo[ks].u[i]);
            }
        }
    }

    float best[8];
    int   bidx[8];
#pragma unroll
    for (int r = 0; r < 8; ++r) { best[r] = __builtin_inff(); bidx[r] = 0; }

    for (int chunk = 0; chunk < MM; chunk += 64) {
        __syncthreads();
#pragma unroll
        for (int i = 0; i < 8; ++i) {
            int f    = t + (i << 8);
            int prow = f >> 5;
            int pc4  = f & 31;
            const float4 x4 = ((const float4*)(Pb + (size_t)(chunk + prow) * DD))[pc4];
            unsigned short h0,h1,h2,h3,l0,l1,l2,l3;
            bf16_split(x4.x, h0, l0); bf16_split(x4.y, h1, l1);
            bf16_split(x4.z, h2, l2); bf16_split(x4.w, h3, l3);
            int woff = prow * (LSTR / 2) + (pc4 << 1);   // uint index (LSTR even)
            ((unsigned int*)ldsHi)[woff + 0] = (unsigned)h0 | ((unsigned)h1 << 16);
            ((unsigned int*)ldsHi)[woff + 1] = (unsigned)h2 | ((unsigned)h3 << 16);
            ((unsigned int*)ldsLo)[woff + 0] = (unsigned)l0 | ((unsigned)l1 << 16);
            ((unsigned int*)ldsLo)[woff + 1] = (unsigned)l2 | ((unsigned)l3 << 16);
        }
        if (t < 64) {   // deterministic fixed-order row sum (L2-hot re-read)
            const float4* rp = (const float4*)(Pb + (size_t)(chunk + t) * DD);
            float s = 0.0f;
#pragma unroll
            for (int j = 0; j < DD / 4; ++j) {
                float4 v = rp[j];
                s += v.x*v.x + v.y*v.y + v.z*v.z + v.w*v.w;
            }
            ldsP2[t] = s;
        }
        __syncthreads();

#pragma unroll
        for (int st = 0; st < 4; ++st) {
            const int col  = (st << 4) + row16;
            const int koff = col * LSTR + (half << 4);
            v8f c0 = {0.f,0.f,0.f,0.f,0.f,0.f,0.f,0.f};
            v8f c1 = {0.f,0.f,0.f,0.f,0.f,0.f,0.f,0.f};
#pragma unroll
            for (int ks = 0; ks < 4; ++ks) {
                V16BF bh, bl;
                load_frag(&ldsHi[koff + (ks << 5)], bh);
                load_frag(&ldsLo[koff + (ks << 5)], bl);
                if (ks & 1) {
                    c1 = __builtin_amdgcn_wmma_f32_16x16x32_bf16(false, ahi[ks].v, false, bh.v,
                                                                 (short)0, c1, false, false);
                    c1 = __builtin_amdgcn_wmma_f32_16x16x32_bf16(false, ahi[ks].v, false, bl.v,
                                                                 (short)0, c1, false, false);
                    c1 = __builtin_amdgcn_wmma_f32_16x16x32_bf16(false, alo[ks].v, false, bh.v,
                                                                 (short)0, c1, false, false);
                } else {
                    c0 = __builtin_amdgcn_wmma_f32_16x16x32_bf16(false, ahi[ks].v, false, bh.v,
                                                                 (short)0, c0, false, false);
                    c0 = __builtin_amdgcn_wmma_f32_16x16x32_bf16(false, ahi[ks].v, false, bl.v,
                                                                 (short)0, c0, false, false);
                    c0 = __builtin_amdgcn_wmma_f32_16x16x32_bf16(false, alo[ks].v, false, bh.v,
                                                                 (short)0, c0, false, false);
                }
            }
            const float p2c  = ldsP2[(st << 4) + row16];
            const int   mIdx = chunk + (st << 4) + row16;
#pragma unroll
            for (int r = 0; r < 8; ++r) {
                float s = __builtin_fmaf(-2.0f, c0[r] + c1[r], p2c);
                if (s < best[r]) { best[r] = s; bidx[r] = mIdx; }
            }
        }
    }

#pragma unroll
    for (int r = 0; r < 8; ++r) {
        float bv = best[r]; int bi = bidx[r];
#pragma unroll
        for (int m = 1; m <= 8; m <<= 1) {
            float ov = __shfl_xor(bv, m, 32);
            int   oi = __shfl_xor(bi, m, 32);
            if (ov < bv || (ov == bv && oi < bi)) { bv = ov; bi = oi; }
        }
        best[r] = bv; bidx[r] = bi;
    }
    if (row16 == 0) {
#pragma unroll
        for (int r = 0; r < 8; ++r)
            ldsIdx[wave * 16 + (half << 3) + r] = bidx[r];
    }
    __syncthreads();

    {
        const int q = t >> 1;
        const int h = t & 1;
        const int idx = ldsIdx[q];
        const float4* src = (const float4*)(Pb + (size_t)idx * DD + h * 64);
        float4* dst = (float4*)(out + ((size_t)b * NN + qBaseWG + q) * DD + h * 64);
#pragma unroll
        for (int j = 0; j < 16; ++j) dst[j] = src[j];
    }
}

extern "C" void kernel_launch(void* const* d_in, const int* in_sizes, int n_in,
                              void* d_out, int out_size, void* d_ws, size_t ws_size,
                              hipStream_t stream) {
    (void)in_sizes; (void)n_in; (void)out_size;
    const float* P = (const float*)d_in[0];   // prompt_feats [B,M,D]
    const float* Q = (const float*)d_in[1];   // query_feats  [B,N,D]
    float* out = (float*)d_out;

    const size_t elems     = (size_t)BB * MM * DD;        // 4,194,304
    const size_t phi_bytes = elems * sizeof(unsigned short);
    const size_t p2_bytes  = (size_t)BB * MM * sizeof(float);
    const size_t need      = 2 * phi_bytes + p2_bytes;    // ~16.9 MB

    dim3 grid(NN / 128, BB);   // 32 x 8 workgroups, 256 threads each

    if (d_ws != nullptr && ws_size >= need) {
        unsigned short* Phi = (unsigned short*)d_ws;
        unsigned short* Plo = Phi + elems;
        float* P2 = (float*)((char*)d_ws + 2 * phi_bytes);
        hipLaunchKernelGGL(split_kernel, dim3((unsigned)(elems / 4 / 256)), dim3(256),
                           0, stream, P, Phi, Plo);
        hipLaunchKernelGGL(p2_kernel, dim3((unsigned)(BB * MM / 256)), dim3(256),
                           0, stream, P, P2);
        hipLaunchKernelGGL(nn_wmma_pre_kernel, grid, dim3(256), 0, stream,
                           Phi, Plo, P2, P, Q, out);
    } else {
        hipLaunchKernelGGL(nn_gather_kernel, grid, dim3(256), 0, stream, P, Q, out);
    }
}
